// STICKYLlamaAttention_2405181686051
// MI455X (gfx1250) — compile-verified, matrix operand order checked by reading, not run on hardware
//
#include <hip/hip_runtime.h>

#define S_LEN 2048
#define HID   4096
#define NH    32
#define NKV   8
#define HD    128
#define QD    (NH * HD)     // 4096
#define KVD   (NKV * HD)    // 1024
#define SM_SCALE 0.08838834764831845f   // 1/sqrt(128)

typedef __attribute__((ext_vector_type(16))) __bf16 v16bf;
typedef __attribute__((ext_vector_type(8)))  float  v8f;

// Load a 16-element bf16 WMMA fragment (A or B operand) from an LDS row.
// Per-lane packing (16-bit 16x32 operand): elements kb..kb+7 and 16+kb..16+kb+7,
// where kb = (lane>>4)*8. Two 16-byte DS loads.
static __device__ __forceinline__ v16bf load_frag(const __bf16* row, int kb) {
  v16bf f;
  ((float4*)&f)[0] = *(const float4*)(row + kb);
  ((float4*)&f)[1] = *(const float4*)(row + 16 + kb);
  return f;
}

// CDNA5 async global->LDS copy (16B per lane). LDS operand is the wave-relative
// byte offset (low 32 bits of the shared-memory pointer). Tracked on ASYNCcnt.
static __device__ __forceinline__ void async_copy_b128(const void* lds_dst,
                                                       const void* gsrc) {
  unsigned lds_off = (unsigned)(unsigned long long)lds_dst;
  unsigned long long ga = (unsigned long long)gsrc;
  asm volatile("global_load_async_to_lds_b128 %0, %1, off"
               :: "v"(lds_off), "v"(ga)
               : "memory");
}
static __device__ __forceinline__ void wait_async() {
  asm volatile("s_wait_asynccnt 0x0" ::: "memory");
}

// ---------------------------------------------------------------------------
// C[M,N] (f32) = A[M,K] (bf16, row-major) @ W[N,K]^T  (W fp32, converted to
// bf16 on the fly while staging tiles into LDS).
// Block: 256 threads = 8 waves (4 in M x 2 in N). Tile: BM=128, BN=128, BK=32.
// Wave computes a 32x64 strip => 8 WMMAs per K-step reusing 4 B-frags.
// A-tile staged via async global->LDS (pure bf16 copy).
// ---------------------------------------------------------------------------
__global__ __launch_bounds__(256) void gemm_bf16_xwt(
    const __bf16* __restrict__ A, const float* __restrict__ W,
    float* __restrict__ C, int M, int N, int K)
{
  __shared__ __bf16 sA[128][32];   // 8 KB
  __shared__ __bf16 sB[128][32];   // 8 KB

  const int tid  = threadIdx.x;
  const int wave = tid >> 5, lane = tid & 31;
  const int fr   = lane & 15;
  const int kb   = (lane >> 4) << 3;
  const int hi   = lane >> 4;
  const int wm   = (wave & 3) * 32;    // wave row offset within tile
  const int wn   = (wave >> 2) * 64;   // wave col offset within tile
  const int m0   = blockIdx.x * 128, n0 = blockIdx.y * 128;

  v8f acc[2][4] = {};

  const int ra = (tid * 16) >> 5, ca = (tid * 16) & 31;  // A: 16 bf16/thread
  const int rb = ra, cb = ca;                            // B: 16 f32/thread

  for (int k0 = 0; k0 < K; k0 += 32) {
    // A tile: two async 16B copies per thread (no VGPR round-trip)
    const __bf16* asrc = A + (size_t)(m0 + ra) * K + k0 + ca;
    async_copy_b128(&sA[ra][ca],     asrc);
    async_copy_b128(&sA[ra][ca + 8], asrc + 8);

    // B tile: load 16 f32, convert to bf16, two 16B DS stores
    const float* wsrc = W + (size_t)(n0 + rb) * K + k0 + cb;
    float4 w0 = ((const float4*)wsrc)[0];
    float4 w1 = ((const float4*)wsrc)[1];
    float4 w2 = ((const float4*)wsrc)[2];
    float4 w3 = ((const float4*)wsrc)[3];
    union { __bf16 b[8]; float4 f; } c0, c1;
    c0.b[0] = (__bf16)w0.x; c0.b[1] = (__bf16)w0.y;
    c0.b[2] = (__bf16)w0.z; c0.b[3] = (__bf16)w0.w;
    c0.b[4] = (__bf16)w1.x; c0.b[5] = (__bf16)w1.y;
    c0.b[6] = (__bf16)w1.z; c0.b[7] = (__bf16)w1.w;
    c1.b[0] = (__bf16)w2.x; c1.b[1] = (__bf16)w2.y;
    c1.b[2] = (__bf16)w2.z; c1.b[3] = (__bf16)w2.w;
    c1.b[4] = (__bf16)w3.x; c1.b[5] = (__bf16)w3.y;
    c1.b[6] = (__bf16)w3.z; c1.b[7] = (__bf16)w3.w;
    *(float4*)&sB[rb][cb]     = c0.f;
    *(float4*)&sB[rb][cb + 8] = c1.f;

    wait_async();
    __syncthreads();

    v16bf af0 = load_frag(&sA[wm + fr][0],      kb);
    v16bf af1 = load_frag(&sA[wm + 16 + fr][0], kb);
#pragma unroll
    for (int j = 0; j < 4; ++j) {
      v16bf bf = load_frag(&sB[wn + j * 16 + fr][0], kb);
      acc[0][j] = __builtin_amdgcn_wmma_f32_16x16x32_bf16(
          false, af0, false, bf, (short)0, acc[0][j], false, false);
      acc[1][j] = __builtin_amdgcn_wmma_f32_16x16x32_bf16(
          false, af1, false, bf, (short)0, acc[1][j], false, false);
    }
    __syncthreads();
  }

  // C/D layout: VGPR r, lane: row = hi*8 + r, col = lane%16
#pragma unroll
  for (int a = 0; a < 2; ++a)
#pragma unroll
    for (int j = 0; j < 4; ++j)
#pragma unroll
      for (int r = 0; r < 8; ++r)
        C[(size_t)(m0 + wm + a * 16 + hi * 8 + r) * N + n0 + wn + j * 16 + fr]
            = acc[a][j][r];
}

// ---------------------------------------------------------------------------
// llama3-scaled RoPE applied to fp32 projections, output bf16.
// X,Y layout: [S, nheads*128]; one thread handles one (s, head, i<64) pair.
// ---------------------------------------------------------------------------
__global__ void rope_to_bf16(const float* __restrict__ X, __bf16* __restrict__ Y,
                             int nheads, int total)
{
  int idx = blockIdx.x * blockDim.x + threadIdx.x;
  if (idx >= total) return;
  int i  = idx & 63;
  int t  = idx >> 6;
  int hh = t % nheads;
  int s  = t / nheads;

  float inv     = powf(500000.0f, -(float)(2 * i) / 128.0f);
  float wavelen = 6.283185307179586f / inv;
  float smooth  = (8192.0f / wavelen - 1.0f) * (1.0f / 3.0f);
  smooth        = fminf(fmaxf(smooth, 0.0f), 1.0f);
  float nif     = (1.0f - smooth) * (inv * 0.125f) + smooth * inv;
  float ang     = (float)s * nif;
  float c = cosf(ang), sn = sinf(ang);

  size_t base = (size_t)s * nheads * 128 + (size_t)hh * 128;
  float x1 = X[base + i], x2 = X[base + i + 64];
  Y[base + i]      = (__bf16)(x1 * c - x2 * sn);
  Y[base + i + 64] = (__bf16)(x2 * c + x1 * sn);
}

__global__ void f32_to_bf16(const float* __restrict__ X, __bf16* __restrict__ Y, int n) {
  int i = blockIdx.x * blockDim.x + threadIdx.x;
  if (i < n) Y[i] = (__bf16)X[i];
}

// ---------------------------------------------------------------------------
// Flash attention: grid (head, q-tile). 128 threads = 4 waves; each wave owns
// a 16-row q slice. Q/K tiles staged with async global->LDS; V is transposed
// through VGPRs so P@V B-frags are two contiguous 16B DS loads. Online softmax
// with 16-lane shfl_xor reductions (rows live in half-waves in C/D layout).
// ---------------------------------------------------------------------------
__global__ __launch_bounds__(128) void flash_attn_gqa(
    const __bf16* __restrict__ Q, const __bf16* __restrict__ Kmat,
    const __bf16* __restrict__ Vmat, __bf16* __restrict__ O)
{
  __shared__ __bf16 sQ[64][128];     // 16 KB
  __shared__ __bf16 sK[64][128];     // 16 KB
  __shared__ __bf16 sVt[128][64];    // 16 KB (V transposed: [dim][kpos])
  __shared__ __bf16 sP[4][16][64];   //  8 KB (per-wave P tiles)

  const int h   = blockIdx.x;
  const int qt  = blockIdx.y;
  const int hkv = h >> 2;            // N_GROUPS = 4
  const int tid = threadIdx.x, wave = tid >> 5, lane = tid & 31;
  const int fr  = lane & 15, kb = (lane >> 4) << 3, hi = lane >> 4;
  const int q0  = qt * 64;

  // Stage Q tile (64 x 128 bf16) asynchronously
  for (int i = tid; i < 1024; i += 128) {
    int r = i >> 4, c = (i & 15) << 3;
    async_copy_b128(&sQ[r][c], Q + (size_t)(q0 + r) * QD + h * HD + c);
  }
  wait_async();
  __syncthreads();

  v16bf qf[4];
#pragma unroll
  for (int s = 0; s < 4; ++s) qf[s] = load_frag(&sQ[wave * 16 + fr][s * 32], kb);

  float m_run[8], l_run[8];
#pragma unroll
  for (int r = 0; r < 8; ++r) { m_run[r] = -3.0e38f; l_run[r] = 0.0f; }
  v8f o_acc[8] = {};

  for (int kt = 0; kt <= qt; ++kt) {
    const int k0 = kt * 64;
    __syncthreads();   // previous tile's consumers done before overwrite
    for (int i = tid; i < 1024; i += 128) {
      int r = i >> 4, c = (i & 15) << 3;
      async_copy_b128(&sK[r][c],
                      Kmat + (size_t)(k0 + r) * KVD + hkv * HD + c);
      float4 vv = *(const float4*)(Vmat + (size_t)(k0 + r) * KVD + hkv * HD + c);
      const __bf16* pv = (const __bf16*)&vv;
#pragma unroll
      for (int q = 0; q < 8; ++q) sVt[c + q][r] = pv[q];
    }
    wait_async();
    __syncthreads();

    // S = Q K^T : per wave 16x64, K-dim 128 in four 32-steps
    v8f s_acc[4] = {};
#pragma unroll
    for (int s = 0; s < 4; ++s) {
#pragma unroll
      for (int j = 0; j < 4; ++j) {
        v16bf kf = load_frag(&sK[j * 16 + fr][s * 32], kb);
        s_acc[j] = __builtin_amdgcn_wmma_f32_16x16x32_bf16(
            false, qf[s], false, kf, (short)0, s_acc[j], false, false);
      }
    }

    // Scale + causal mask
#pragma unroll
    for (int j = 0; j < 4; ++j) {
      int col = k0 + j * 16 + fr;
#pragma unroll
      for (int r = 0; r < 8; ++r) {
        int row = q0 + wave * 16 + hi * 8 + r;
        float v = s_acc[j][r] * SM_SCALE;
        s_acc[j][r] = (col > row) ? -3.0e38f : v;
      }
    }

    // Online softmax: rows replicated across 16 lanes of each half-wave
#pragma unroll
    for (int r = 0; r < 8; ++r) {
      float mr = fmaxf(fmaxf(s_acc[0][r], s_acc[1][r]),
                       fmaxf(s_acc[2][r], s_acc[3][r]));
#pragma unroll
      for (int off = 1; off < 16; off <<= 1)
        mr = fmaxf(mr, __shfl_xor(mr, off, 32));
      float mnew  = fmaxf(m_run[r], mr);
      float alpha = __expf(m_run[r] - mnew);
      m_run[r] = mnew;
      float rs = 0.0f;
#pragma unroll
      for (int j = 0; j < 4; ++j) {
        float p = __expf(s_acc[j][r] - mnew);
        s_acc[j][r] = p;
        rs += p;
      }
#pragma unroll
      for (int off = 1; off < 16; off <<= 1)
        rs += __shfl_xor(rs, off, 32);
      l_run[r] = l_run[r] * alpha + rs;
#pragma unroll
      for (int n = 0; n < 8; ++n) o_acc[n][r] *= alpha;
    }

    // Re-layout P (C-frag -> A-frag) via per-wave LDS region
#pragma unroll
    for (int j = 0; j < 4; ++j)
#pragma unroll
      for (int r = 0; r < 8; ++r)
        sP[wave][hi * 8 + r][j * 16 + fr] = (__bf16)s_acc[j][r];

    // O += P @ V  (K-dim 64 = two 32-steps, N = 128 dims = 8 subtiles)
#pragma unroll
    for (int ks = 0; ks < 2; ++ks) {
      v16bf pf = load_frag(&sP[wave][fr][ks * 32], kb);
#pragma unroll
      for (int n = 0; n < 8; ++n) {
        v16bf vf = load_frag(&sVt[n * 16 + fr][ks * 32], kb);
        o_acc[n] = __builtin_amdgcn_wmma_f32_16x16x32_bf16(
            false, pf, false, vf, (short)0, o_acc[n], false, false);
      }
    }
  }

  // Normalize and write attention output (bf16) for the O-projection GEMM
#pragma unroll
  for (int n = 0; n < 8; ++n)
#pragma unroll
    for (int r = 0; r < 8; ++r) {
      int row = q0 + wave * 16 + hi * 8 + r;
      float o = o_acc[n][r] / l_run[r];
      O[(size_t)row * QD + h * HD + n * 16 + fr] = (__bf16)o;
    }
}

// ---------------------------------------------------------------------------
// Orchestration. Workspace layout (bytes):
//   [0,16M)    Xbf   : hidden_states bf16         [2048,4096]
//   [16M,48M)  Qf32  : Q projection fp32          [2048,4096]  (recycled: Obf)
//   [48M,56M)  Kf32  : K projection fp32          [2048,1024]
//   [56M,64M)  Vf32  : V projection fp32          [2048,1024]
//   [64M,80M)  Qbf   : Q post-RoPE bf16
//   [80M,84M)  Kbf   : K post-RoPE bf16
//   [84M,88M)  Vbf   : V bf16
// ---------------------------------------------------------------------------
extern "C" void kernel_launch(void* const* d_in, const int* in_sizes, int n_in,
                              void* d_out, int out_size, void* d_ws, size_t ws_size,
                              hipStream_t stream) {
  const float* hs = (const float*)d_in[0];
  const float* Wq = (const float*)d_in[1];
  const float* Wk = (const float*)d_in[2];
  const float* Wv = (const float*)d_in[3];
  const float* Wo = (const float*)d_in[4];
  float* out = (float*)d_out;

  char* ws = (char*)d_ws;
  const size_t MB = 1024ull * 1024ull;
  __bf16* Xbf = (__bf16*)(ws);
  float*  Qf  = (float*)(ws + 16 * MB);
  float*  Kf  = (float*)(ws + 48 * MB);
  float*  Vf  = (float*)(ws + 56 * MB);
  __bf16* Qbf = (__bf16*)(ws + 64 * MB);
  __bf16* Kbf = (__bf16*)(ws + 80 * MB);
  __bf16* Vbf = (__bf16*)(ws + 84 * MB);
  __bf16* Obf = (__bf16*)(ws + 16 * MB);   // reuse Qf region

  // 1. hidden states -> bf16
  f32_to_bf16<<<(S_LEN * HID) / 256, 256, 0, stream>>>(hs, Xbf, S_LEN * HID);

  // 2. projections (WMMA GEMMs, f32 accumulate)
  gemm_bf16_xwt<<<dim3(S_LEN / 128, QD / 128),  256, 0, stream>>>(Xbf, Wq, Qf, S_LEN, QD,  HID);
  gemm_bf16_xwt<<<dim3(S_LEN / 128, KVD / 128), 256, 0, stream>>>(Xbf, Wk, Kf, S_LEN, KVD, HID);
  gemm_bf16_xwt<<<dim3(S_LEN / 128, KVD / 128), 256, 0, stream>>>(Xbf, Wv, Vf, S_LEN, KVD, HID);

  // 3. RoPE (Q,K) + bf16 casts
  rope_to_bf16<<<(S_LEN * NH * 64) / 256, 256, 0, stream>>>(Qf, Qbf, NH, S_LEN * NH * 64);
  rope_to_bf16<<<(S_LEN * NKV * 64) / 256, 256, 0, stream>>>(Kf, Kbf, NKV, S_LEN * NKV * 64);
  f32_to_bf16<<<(S_LEN * KVD) / 256, 256, 0, stream>>>(Vf, Vbf, S_LEN * KVD);

  // 4. flash attention (causal, GQA 4:1)
  flash_attn_gqa<<<dim3(NH, S_LEN / 64), 128, 0, stream>>>(Qbf, Kbf, Vbf, Obf);

  // 5. output projection -> fp32 d_out
  gemm_bf16_xwt<<<dim3(S_LEN / 128, HID / 128), 256, 0, stream>>>(Obf, Wo, out, S_LEN, HID, QD);
}
